// RNN_3427383902243
// MI455X (gfx1250) — matrix operand order repaired and checked by reading, not compiled, not run on hardware
//
#include <hip/hip_runtime.h>
#include <hip/hip_bf16.h>
#include <stdint.h>

// ---------------------------------------------------------------------------
// 2-layer ReLU RNN on MI455X (gfx1250, wave32, WMMA + TDM).
//
// Pipeline (all on `stream`, graph-capture safe):
//   1. pack_w      : fp32 weights -> bf16, pre-swizzled into WMMA B-fragment order
//   2. cvt_x       : x [B,T,I] fp32 -> [T,B,I] bf16 (A operand of U0 GEMM)
//   3. gemm        : S0 = x_bf @ W_ih0^T          (v_wmma_f32_16x16x32_bf16)
//   4. rnn_scan    : S0[t] = relu(S0[t] + h_{t-1} @ W_hh0^T + b), persistent
//                    kernel, per-step grid barrier. W_hh tile staged to LDS via
//                    TDM (tensor_load_to_lds) then register-resident B frags.
//   5. gemm        : S1 = h0_bf @ W_ih1^T
//   6. rnn_scan    : layer 1, same scheme
//   7. gemm + bias : out = h1_last_bf @ fc_w^T + fc_b
//
// Workspace requirement: ~673 MB
// ---------------------------------------------------------------------------

#define B_ 128
#define T_ 1024
#define I_ 256
#define H_ 512
#define C_ 128

typedef __attribute__((ext_vector_type(16))) __bf16 v16bf;
typedef __attribute__((ext_vector_type(8)))  float  v8f;
typedef __attribute__((ext_vector_type(4)))  unsigned int u32x4;
typedef __attribute__((ext_vector_type(8)))  int   i32x8;
typedef __attribute__((ext_vector_type(4)))  int   i32x4;

#if defined(__has_builtin)
#  if __has_builtin(__builtin_amdgcn_tensor_load_to_lds) && \
      __has_builtin(__builtin_amdgcn_s_wait_tensorcnt)
#    define TDM_AVAILABLE 1
#  else
#    define TDM_AVAILABLE 0
#  endif
#else
#  define TDM_AVAILABLE 0
#endif

union FragB16 { uint4 u[2]; v16bf v; };   // 32 bytes = one 16-bf16 WMMA operand

__device__ __forceinline__ uint32_t f2bf_pk(float lo, float hi) {
  uint32_t a = __float_as_uint(lo);
  uint32_t b = __float_as_uint(hi);
  a = (a + 0x7FFFu + ((a >> 16) & 1u)) >> 16;   // round-to-nearest-even
  b = (b + 0x7FFFu + ((b >> 16) & 1u)) >> 16;
  return a | (b << 16);
}
__device__ __forceinline__ uint16_t f2bf(float f) {
  uint32_t a = __float_as_uint(f);
  return (uint16_t)((a + 0x7FFFu + ((a >> 16) & 1u)) >> 16);
}

// --------------------------------------------------------------------------
// Pack W [N,K] fp32 into WMMA B-fragment order (bf16):
// dword index = ((nt*ktiles + kt)*32 + lane)*8 + v
//   n = nt*16 + lane%16 ; k = kt*32 + 2v + 16*(lane/16); dword = {W[n,k],W[n,k+1]}
// --------------------------------------------------------------------------
__global__ void pack_w(const float* __restrict__ W, uint32_t* __restrict__ P,
                       int K, int total) {
  int tid = blockIdx.x * blockDim.x + threadIdx.x;
  if (tid >= total) return;
  int v      = tid & 7;
  int lane   = (tid >> 3) & 31;
  int tile   = tid >> 8;
  int ktiles = K >> 5;
  int kt = tile % ktiles;
  int nt = tile / ktiles;
  int n  = nt * 16 + (lane & 15);
  int k  = kt * 32 + 2 * v + 16 * (lane >> 4);
  const float* w = W + (size_t)n * K + k;
  P[tid] = f2bf_pk(w[0], w[1]);
}

// x [B,T,I] fp32 -> Abf [T,B,I] bf16, two elements per thread.
__global__ void cvt_x(const float* __restrict__ x, uint16_t* __restrict__ o) {
  size_t tid = (size_t)blockIdx.x * blockDim.x + threadIdx.x;  // T*B*I/2 threads
  int    i2  = (int)(tid & (I_ / 2 - 1));                      // I/2 = 128
  size_t tb  = tid >> 7;
  int    b   = (int)(tb & (B_ - 1));
  int    t   = (int)(tb >> 7);
  const float* src = x + ((size_t)b * T_ + t) * I_ + i2 * 2;
  uint32_t val = f2bf_pk(src[0], src[1]);
  *(uint32_t*)(o + ((size_t)t * B_ + b) * I_ + i2 * 2) = val;
}

// --------------------------------------------------------------------------
// C[M,N] = A[M,K](bf16, row-major) @ Wp^T (packed bf16). No bias.
// grid = (M/128, N/64), block = 256 (8 waves). Each wave: 16 rows x 64 cols.
// --------------------------------------------------------------------------
__global__ __launch_bounds__(256) void gemm_bf16_wmma(
    const uint16_t* __restrict__ A, const uint32_t* __restrict__ Wp,
    float* __restrict__ C, int K, int N) {
  const int lane = threadIdx.x & 31;
  const int wave = threadIdx.x >> 5;
  const int nbase = blockIdx.y * 64;
  const int ktiles = K >> 5;
  const size_t mtile = (size_t)blockIdx.x * 8 + wave;     // 16-row tile index
  const size_t mrow  = mtile * 16 + (lane & 15);          // A row for this lane
  const uint16_t* arow = A + mrow * (size_t)K + 16 * (lane >> 4);

  v8f z = {0.f, 0.f, 0.f, 0.f, 0.f, 0.f, 0.f, 0.f};
  v8f acc[4] = {z, z, z, z};

  for (int kt = 0; kt < ktiles; ++kt) {
    FragB16 a;
    a.u[0] = *(const uint4*)(arow);
    a.u[1] = *(const uint4*)(arow + 8);
    arow += 32;
    __builtin_prefetch(arow, 0, 1);  // global_prefetch_b8: next K slice
#pragma unroll
    for (int j = 0; j < 4; ++j) {
      const size_t nt = (size_t)(nbase >> 4) + j;
      FragB16 b;
      const uint4* bp = (const uint4*)(Wp + ((nt * ktiles + kt) * 32 + lane) * 8);
      b.u[0] = bp[0];
      b.u[1] = bp[1];
      acc[j] = __builtin_amdgcn_wmma_f32_16x16x32_bf16(
          false, a.v, false, b.v, (short)0, acc[j], false, false);
    }
  }
  // C/D layout: VGPR v -> M = v + 8*(lane/16); N = lane%16
  const size_t mbase = mtile * 16 + 8 * (lane >> 4);
  const int nl = lane & 15;
#pragma unroll
  for (int j = 0; j < 4; ++j)
#pragma unroll
    for (int v = 0; v < 8; ++v)
      C[(mbase + v) * (size_t)N + nbase + j * 16 + nl] = acc[j][v];
}

// --------------------------------------------------------------------------
// Persistent recurrent scan: for t in 0..T-1:
//   S[t] = relu(S[t] + h_{t-1} @ Whh^T + b_ih + b_hh);  Hb[t] = bf16(S[t])
// Grid = 32 blocks (one hidden-tile nt each) x 8 waves (one batch-tile mt each).
// All 8 waves of a block share one 16 KB packed-Whh column tile: staged once
// into LDS by the Tensor Data Mover, then lifted into registers (16 B-frags =
// 128 VGPRs) and reused for all T steps. Inner step = 2 global_load_b128 (h
// mirror) + 1 WMMA per k-tile. Per-timestep grid sync via monotonic atomic.
// --------------------------------------------------------------------------
__global__ __launch_bounds__(256) void rnn_scan_wmma(
    float* __restrict__ S, uint16_t* __restrict__ Hb,
    const uint32_t* __restrict__ Wp,
    const float* __restrict__ b_ih, const float* __restrict__ b_hh,
    unsigned int* __restrict__ bar, int T) {
  __shared__ __align__(16) uint32_t wsh[16 * 32 * 8];   // 16 KB packed Whh tile
  const int lane = threadIdx.x & 31;
  const int mt   = threadIdx.x >> 5;   // wave id == batch tile (8 x 16 = 128)
  const int nt   = blockIdx.x;         // hidden tile          (32 x 16 = 512)
  const int nl   = lane & 15;
  const int hh   = lane >> 4;
  const uint32_t* wsrc = Wp + (size_t)nt * (16 * 32 * 8);

#if TDM_AVAILABLE
  if (threadIdx.x == 0) {
    // 1-row TDM copy: 16 KB = 2048 x 8-byte elements, global -> LDS offset 0.
    uint64_t ga = (uint64_t)(uintptr_t)wsrc;
    u32x4 g0 = { 1u,                                   // count=1 valid descriptor
                 0u,                                   // lds_addr (wsh is at 0)
                 (uint32_t)ga,                         // global_addr[31:0]
                 (uint32_t)(((ga >> 32) & 0x01FFFFFFu) // global_addr[56:32]
                            | (2u << 30)) };           // type = 2 ("image")
    i32x8 g1 = { (int)(3u << 16),        // workgroup_mask=0, data_size=3 (8B)
                 (int)(2048u << 16),     // tensor_dim0 = 2048 (lo16)
                 (int)(1u << 16),        // tensor_dim0 hi=0, tensor_dim1 = 1
                 (int)(2048u << 16),     // tensor_dim1 hi=0, tile_dim0 = 2048
                 1,                      // tile_dim1 = 1, tile_dim2 = 0
                 2048,                   // tensor_dim0_stride = 2048
                 0, 0 };
    i32x4 gz = { 0, 0, 0, 0 };
#if __clang_major__ >= 23
    i32x8 g4 = { 0, 0, 0, 0, 0, 0, 0, 0 };
    __builtin_amdgcn_tensor_load_to_lds(g0, g1, gz, gz, g4, 0);
#else
    __builtin_amdgcn_tensor_load_to_lds(g0, g1, gz, gz, 0);
#endif
    __builtin_amdgcn_s_wait_tensorcnt(0);
  }
  // Unprovable-dead store: stops the compiler assuming wsh is uninitialized
  // (the TDM engine wrote it behind its back).
  if (blockDim.y == 12345u) wsh[0] = 0u;
#else
  for (int i = threadIdx.x; i < 16 * 32 * 8; i += 256) wsh[i] = wsrc[i];
#endif
  __syncthreads();

  // Lift this wave's 16 B-fragments LDS -> registers (held for all T steps).
  FragB16 bfr[16];
#pragma unroll
  for (int kt = 0; kt < 16; ++kt) {
    const uint4* lp = (const uint4*)(wsh + (size_t)(kt * 32 + lane) * 8);
    bfr[kt].u[0] = lp[0];
    bfr[kt].u[1] = lp[1];
  }

  const float bias = b_ih[nt * 16 + nl] + b_hh[nt * 16 + nl];
  const unsigned nblocks = gridDim.x;
  const v8f z = {0.f, 0.f, 0.f, 0.f, 0.f, 0.f, 0.f, 0.f};

  for (int t = 0; t < T; ++t) {
    v8f acc = z;
    if (t > 0) {
      // A fragments from bf16 mirror of h_{t-1} (L2-resident, 128 KB)
      const uint16_t* hp = Hb + (size_t)(t - 1) * (B_ * H_)
                              + (size_t)(mt * 16 + nl) * H_ + 16 * hh;
#pragma unroll
      for (int kt = 0; kt < 16; ++kt) {
        FragB16 a;
        a.u[0] = *(const uint4*)(hp);
        a.u[1] = *(const uint4*)(hp + 8);
        hp += 32;
        acc = __builtin_amdgcn_wmma_f32_16x16x32_bf16(
            false, a.v, false, bfr[kt].v, (short)0, acc, false, false);
      }
    }
    // Epilogue: add U (already in S[t]) + bias, ReLU, write fp32 + bf16 mirror
    float*    srow = S  + (size_t)t * (B_ * H_);
    uint16_t* hrow = Hb + (size_t)t * (B_ * H_);
    const int n = nt * 16 + nl;
#pragma unroll
    for (int v = 0; v < 8; ++v) {
      const size_t idx = (size_t)(mt * 16 + 8 * hh + v) * H_ + n;
      float r = fmaxf(acc[v] + srow[idx] + bias, 0.f);
      srow[idx] = r;
      hrow[idx] = f2bf(r);
    }
    // ---- grid-wide barrier (monotonic counter; zeroed per launch) ----
    __syncthreads();
    if (threadIdx.x == 0) {
      __threadfence();
      atomicAdd(bar, 1u);
      const unsigned target = nblocks * (unsigned)(t + 1);
      while (atomicAdd(bar, 0u) < target) __builtin_amdgcn_s_sleep(1);
    }
    __syncthreads();
  }
}

__global__ void bias_add(float* __restrict__ out, const float* __restrict__ b,
                         int N, int total) {
  int i = blockIdx.x * blockDim.x + threadIdx.x;
  if (i < total) out[i] += b[i % N];
}

// --------------------------------------------------------------------------
extern "C" void kernel_launch(void* const* d_in, const int* in_sizes, int n_in,
                              void* d_out, int out_size, void* d_ws, size_t ws_size,
                              hipStream_t stream) {
  const float* x    = (const float*)d_in[0];
  const float* wih0 = (const float*)d_in[1];
  const float* whh0 = (const float*)d_in[2];
  const float* bih0 = (const float*)d_in[3];
  const float* bhh0 = (const float*)d_in[4];
  const float* wih1 = (const float*)d_in[5];
  const float* whh1 = (const float*)d_in[6];
  const float* bih1 = (const float*)d_in[7];
  const float* bhh1 = (const float*)d_in[8];
  const float* fcw  = (const float*)d_in[9];
  const float* fcb  = (const float*)d_in[10];
  float* out = (float*)d_out;

  const size_t TBH = (size_t)T_ * B_ * H_;   // 67,108,864 elements
  char* ws = (char*)d_ws;
  float*    S0    = (float*)(ws);                  // U0 -> h0 (in place), fp32
  float*    S1    = (float*)(ws + TBH * 4);        // U1 -> h1 (in place), fp32
  uint16_t* Abf   = (uint16_t*)(ws + TBH * 8);     // bf16 staging/mirror
  uint32_t* Wih0p = (uint32_t*)(ws + TBH * 10);    // packed weights
  const int dw_ih0 = (H_ / 16) * (I_ / 32) * 32 * 8;   // 65536 dwords
  const int dw_hh  = (H_ / 16) * (H_ / 32) * 32 * 8;   // 131072 dwords
  const int dw_fc  = (C_ / 16) * (H_ / 32) * 32 * 8;   // 32768 dwords
  uint32_t* Whh0p = Wih0p + dw_ih0;
  uint32_t* Wih1p = Whh0p + dw_hh;
  uint32_t* Whh1p = Wih1p + dw_hh;
  uint32_t* FcWp  = Whh1p + dw_hh;
  unsigned* bar0  = (unsigned*)(FcWp + dw_fc);
  unsigned* bar1  = bar0 + 64;   // separate cache line

  // Barrier counters must start at 0 every call (capture-safe async memset).
  hipMemsetAsync(bar0, 0, 512, stream);

  // 1) pack weights into WMMA B-fragment order
  pack_w<<<dw_ih0 / 256, 256, 0, stream>>>(wih0, Wih0p, I_, dw_ih0);
  pack_w<<<dw_hh  / 256, 256, 0, stream>>>(whh0, Whh0p, H_, dw_hh);
  pack_w<<<dw_hh  / 256, 256, 0, stream>>>(wih1, Wih1p, H_, dw_hh);
  pack_w<<<dw_hh  / 256, 256, 0, stream>>>(whh1, Whh1p, H_, dw_hh);
  pack_w<<<dw_fc  / 256, 256, 0, stream>>>(fcw,  FcWp,  H_, dw_fc);

  // 2) x [B,T,I] fp32 -> Abf [T,B,I] bf16
  cvt_x<<<(unsigned)((size_t)T_ * B_ * I_ / 2 / 256), 256, 0, stream>>>(x, Abf);

  // 3) U0 = x_bf @ Wih0^T      (M = T*B = 131072, K = 256, N = 512)
  dim3 gU((unsigned)(TBH / H_ / 128), H_ / 64);
  gemm_bf16_wmma<<<gU, 256, 0, stream>>>(Abf, Wih0p, S0, I_, H_);

  // 4) layer-0 recurrence; writes h0 into S0 and bf16 mirror into Abf
  rnn_scan_wmma<<<32, 256, 0, stream>>>(S0, Abf, Whh0p, bih0, bhh0, bar0, T_);

  // 5) U1 = h0_bf @ Wih1^T     (K = 512)
  gemm_bf16_wmma<<<gU, 256, 0, stream>>>(Abf, Wih1p, S1, H_, H_);

  // 6) layer-1 recurrence; mirror reuses Abf
  rnn_scan_wmma<<<32, 256, 0, stream>>>(S1, Abf, Whh1p, bih1, bhh1, bar1, T_);

  // 7) out = h1_last @ fc_w^T + fc_b   (M = 128, K = 512, N = 128)
  dim3 gF(1, C_ / 64);
  gemm_bf16_wmma<<<gF, 256, 0, stream>>>(Abf + (size_t)(T_ - 1) * B_ * H_,
                                         FcWp, out, H_, C_);
  bias_add<<<(B_ * C_ + 255) / 256, 256, 0, stream>>>(out, fcb, C_, B_ * C_);
}
